// Qwen2_5_VLVisionBlock_48911087567331
// MI455X (gfx1250) — matrix operand order repaired and checked by reading, not compiled, not run on hardware
//
#include <hip/hip_runtime.h>
#include <hip/hip_fp16.h>
#include <math.h>

typedef __attribute__((ext_vector_type(16))) _Float16 v16h;
typedef __attribute__((ext_vector_type(8)))  _Float16 v8h;
typedef __attribute__((ext_vector_type(8)))  float    v8f;

#define SEQ   4096
#define DIM   1280
#define HEADS 16
#define HD    80
#define HDP   96      // head dim padded to multiple of 32 for wmma K
#define FFN   3420
#define FFNP  3456    // padded to multiple of 128 (block N tile)
#define QKVN  3840

static __device__ __forceinline__ v16h cat8(v8h lo, v8h hi) {
  return __builtin_shufflevector(lo, hi, 0,1,2,3,4,5,6,7,8,9,10,11,12,13,14,15);
}

// Async global->LDS copy of 16 bytes per lane (CDNA5 GLOBAL_LOAD_ASYNC_TO_LDS_B128,
// tracked by ASYNCcnt). VDST operand carries the LDS byte address; a generic
// pointer to __shared__ holds that byte offset in its low 32 bits (ISA 10.2:
// low 32 bits used as LDS address).
static __device__ __forceinline__ void async_b128(const _Float16* gsrc, _Float16* ldst) {
  unsigned int l = (unsigned int)(size_t)ldst;
  asm volatile("global_load_async_to_lds_b128 %0, %1, off"
               :: "v"(l), "v"(gsrc)
               : "memory");
}
static __device__ __forceinline__ void wait_async_4() {
  asm volatile("s_wait_asynccnt 0x4" ::: "memory");
}
static __device__ __forceinline__ void wait_async_0() {
  asm volatile("s_wait_asynccnt 0x0" ::: "memory");
}

// ---------------- f32 -> f16 conversion with zero padding ----------------
__global__ void k_cvt_pad(const float* __restrict__ src, _Float16* __restrict__ dst,
                          int srows, int scols, int drows, int dcols) {
  int idx = blockIdx.x * blockDim.x + threadIdx.x;
  if (idx >= drows * dcols) return;
  int r = idx / dcols, c = idx % dcols;
  float v = (r < srows && c < scols) ? src[(size_t)r * scols + c] : 0.f;
  dst[idx] = (_Float16)v;
}

// ---------------- RMSNorm (f32 in, f16 out) ----------------
__global__ __launch_bounds__(256) void k_rmsnorm(const float* __restrict__ x,
                                                 const float* __restrict__ w,
                                                 _Float16* __restrict__ y) {
  __shared__ float sm[256];
  int row = blockIdx.x;
  const float* xr = x + (size_t)row * DIM;
  float p = 0.f;
  for (int i = threadIdx.x; i < DIM; i += 256) { float v = xr[i]; p += v * v; }
  sm[threadIdx.x] = p;
  __syncthreads();
  for (int s = 128; s > 0; s >>= 1) {
    if (threadIdx.x < s) sm[threadIdx.x] += sm[threadIdx.x + s];
    __syncthreads();
  }
  float scale = rsqrtf(sm[0] * (1.f / DIM) + 1e-6f);
  for (int i = threadIdx.x; i < DIM; i += 256)
    y[(size_t)row * DIM + i] = (_Float16)(xr[i] * scale * w[i]);
}

// ---------------- WMMA GEMM: out[M,N] = A[M,K] @ W[N,K]^T + bias (+resid) ----------------
// 256 threads = 8 waves (2x4). Block tile 128x128, wave tile 64x32 (8 wmma / K-step).
// A/B tiles double-buffered in LDS, filled by async global->LDS b128 copies.
#define LSTR 40  // 32 + 8 halves pad (spreads LDS banks)

template <bool RESID>
__global__ __launch_bounds__(256) void k_gemm(const _Float16* __restrict__ A,
                                              const _Float16* __restrict__ W,
                                              const float* __restrict__ bias, int nbias,
                                              const float* __restrict__ resid,
                                              float* __restrict__ out,
                                              int M, int N, int K) {
  __shared__ _Float16 As[2][128][LSTR];
  __shared__ _Float16 Bs[2][128][LSTR];

  int tid = threadIdx.x;
  int lane = tid & 31, wave = tid >> 5;
  int hlf = lane >> 4, r = lane & 15;
  int wm = wave >> 2, wn = wave & 3;            // 2 x 4 wave grid
  int bm0 = blockIdx.y * 128, bn0 = blockIdx.x * 128;

  // tile copy mapping: 512 chunks of 8 halves (16B); chunk c -> row c>>2, col (c&3)*8
  int c0 = tid, c1 = tid + 256;
  int r0 = c0 >> 2, o0 = (c0 & 3) * 8;
  int r1 = c1 >> 2, o1 = (c1 & 3) * 8;

  v8f acc[4][2] = {};
  int NS = K / 32;

  // prologue: stage tile 0
  async_b128(A + (size_t)(bm0 + r0) * K + o0, &As[0][r0][o0]);
  async_b128(A + (size_t)(bm0 + r1) * K + o1, &As[0][r1][o1]);
  async_b128(W + (size_t)(bn0 + r0) * K + o0, &Bs[0][r0][o0]);
  async_b128(W + (size_t)(bn0 + r1) * K + o1, &Bs[0][r1][o1]);

  for (int s = 0; s < NS; s++) {
    int buf = s & 1;
    if (s + 1 < NS) {
      int k0 = (s + 1) * 32;
      int nb = buf ^ 1;
      async_b128(A + (size_t)(bm0 + r0) * K + k0 + o0, &As[nb][r0][o0]);
      async_b128(A + (size_t)(bm0 + r1) * K + k0 + o1, &As[nb][r1][o1]);
      async_b128(W + (size_t)(bn0 + r0) * K + k0 + o0, &Bs[nb][r0][o0]);
      async_b128(W + (size_t)(bn0 + r1) * K + k0 + o1, &Bs[nb][r1][o1]);
      wait_async_4();   // previous tile's 4 copies done (in-order completion)
    } else {
      wait_async_0();
    }
    __syncthreads();

    v16h a[4], b[2];
#pragma unroll
    for (int i = 0; i < 4; i++) {
      const _Float16* p = &As[buf][wm * 64 + i * 16 + r][0];
      a[i] = cat8(*(const v8h*)(p + hlf * 8), *(const v8h*)(p + 16 + hlf * 8));
    }
#pragma unroll
    for (int j = 0; j < 2; j++) {
      const _Float16* p = &Bs[buf][wn * 32 + j * 16 + r][0];
      b[j] = cat8(*(const v8h*)(p + hlf * 8), *(const v8h*)(p + 16 + hlf * 8));
    }
#pragma unroll
    for (int i = 0; i < 4; i++)
#pragma unroll
      for (int j = 0; j < 2; j++)
        acc[i][j] = __builtin_amdgcn_wmma_f32_16x16x32_f16(
            false, a[i], false, b[j], (short)0, acc[i][j], false, false);

    __syncthreads();  // protect the buffer the next issue overwrites
  }

#pragma unroll
  for (int i = 0; i < 4; i++)
#pragma unroll
    for (int j = 0; j < 2; j++)
#pragma unroll
      for (int v = 0; v < 8; v++) {
        int row = bm0 + wm * 64 + i * 16 + v + 8 * hlf;
        int col = bn0 + wn * 32 + j * 16 + r;
        float val = acc[i][j][v] + (col < nbias ? bias[col] : 0.f);
        if (RESID) val += resid[(size_t)row * N + col];
        out[(size_t)row * N + col] = val;
      }
}

// ---------------- RoPE + repack q/k to [h][s][96] (f16), v to [h][d][s] (f16) ----------------
__global__ __launch_bounds__(256) void k_rope_pack(const float* __restrict__ qkv,
                                                   const float* __restrict__ freqs,
                                                   _Float16* __restrict__ q96,
                                                   _Float16* __restrict__ k96,
                                                   _Float16* __restrict__ vt) {
  int s = blockIdx.x;
  const float* row = qkv + (size_t)s * QKVN;
  for (int idx = threadIdx.x; idx < HEADS * HDP; idx += 256) {
    int h = idx / HDP, d = idx % HDP;
    size_t dst = ((size_t)h * SEQ + s) * HDP + d;
    if (d < HD) {
      float f = freqs[(size_t)s * (HD / 2) + (d % (HD / 2))];
      float c = cosf(f), sn = sinf(f);
      float qv = row[h * HD + d];
      float qr = (d < HD / 2) ? -row[h * HD + d + HD / 2] : row[h * HD + d - HD / 2];
      q96[dst] = (_Float16)(qv * c + qr * sn);
      float kv = row[DIM + h * HD + d];
      float kr = (d < HD / 2) ? -row[DIM + h * HD + d + HD / 2] : row[DIM + h * HD + d - HD / 2];
      k96[dst] = (_Float16)(kv * c + kr * sn);
      vt[((size_t)h * HD + d) * SEQ + s] = (_Float16)row[2 * DIM + h * HD + d];
    } else {
      q96[dst] = (_Float16)0.f;
      k96[dst] = (_Float16)0.f;
    }
  }
}

// ---------------- Flash attention: block = (head, 64-query tile), 4 waves x 16 rows ----------------
__global__ __launch_bounds__(128) void k_flash(const _Float16* __restrict__ q96,
                                               const _Float16* __restrict__ k96,
                                               const _Float16* __restrict__ vt,
                                               const float* __restrict__ mask,
                                               _Float16* __restrict__ attn) {
  __shared__ float    Sbuf[4][16][32];
  __shared__ _Float16 Pbuf[4][16][32];
  __shared__ float    alphaLds[64];
  __shared__ float    lLds[64];

  int lane = threadIdx.x & 31, wave = threadIdx.x >> 5;
  int hlf = lane >> 4, r = lane & 15;
  int h = blockIdx.y;
  int q0 = blockIdx.x * 64 + wave * 16;
  const float scale = 0.11180339887498949f;  // 1/sqrt(80)

  v16h qf[3];
  {
    const _Float16* qrow = q96 + ((size_t)h * SEQ + q0 + r) * HDP;
#pragma unroll
    for (int c = 0; c < 3; c++) {
      const _Float16* p = qrow + c * 32 + hlf * 16;
      qf[c] = cat8(*(const v8h*)p, *(const v8h*)(p + 8));
    }
  }

  v8f O[5] = {};
  float m_i = -INFINITY, l_i = 0.f;  // lanes 0..15: one per Q row

  for (int t = 0; t < SEQ; t += 32) {
#pragma unroll
    for (int nt = 0; nt < 2; nt++) {
      v8f s = {};
      const _Float16* krow = k96 + ((size_t)h * SEQ + t + nt * 16 + r) * HDP + hlf * 16;
#pragma unroll
      for (int c = 0; c < 3; c++) {
        v16h b = *(const v16h*)(krow + c * 32);
        s = __builtin_amdgcn_wmma_f32_16x16x32_f16(false, qf[c], false, b, (short)0, s,
                                                   false, false);
      }
#pragma unroll
      for (int v = 0; v < 8; v++) {
        int M = v + 8 * hlf;
        int key = t + nt * 16 + r;
        Sbuf[wave][M][nt * 16 + r] = s[v] * scale + mask[(size_t)(q0 + M) * SEQ + key];
      }
    }
    __syncthreads();

    if (lane < 16) {
      float mx = m_i;
#pragma unroll
      for (int j = 0; j < 32; j++) mx = fmaxf(mx, Sbuf[wave][lane][j]);
      float al = __expf(m_i - mx);
      float ls = 0.f;
#pragma unroll
      for (int j = 0; j < 32; j++) {
        float p = __expf(Sbuf[wave][lane][j] - mx);
        Pbuf[wave][lane][j] = (_Float16)p;
        ls += p;
      }
      m_i = mx;
      l_i = l_i * al + ls;
      alphaLds[wave * 16 + lane] = al;
    }
    __syncthreads();

#pragma unroll
    for (int v = 0; v < 8; v++) {
      float al = alphaLds[wave * 16 + v + 8 * hlf];
#pragma unroll
      for (int n = 0; n < 5; n++) O[n][v] *= al;
    }

    v16h pf = cat8(*(const v8h*)&Pbuf[wave][r][hlf * 16],
                   *(const v8h*)&Pbuf[wave][r][8 + hlf * 16]);

#pragma unroll
    for (int n = 0; n < 5; n++) {
      const _Float16* vp = vt + ((size_t)h * HD + n * 16 + r) * SEQ + t + hlf * 16;
      v16h vb = *(const v16h*)vp;
      O[n] = __builtin_amdgcn_wmma_f32_16x16x32_f16(false, pf, false, vb, (short)0, O[n],
                                                    false, false);
    }
  }

  if (lane < 16) lLds[wave * 16 + lane] = l_i;
  __syncthreads();
#pragma unroll
  for (int v = 0; v < 8; v++) {
    int M = v + 8 * hlf;
    float inv = 1.f / lLds[wave * 16 + M];
#pragma unroll
    for (int n = 0; n < 5; n++)
      attn[(size_t)(q0 + M) * DIM + h * HD + n * 16 + r] = (_Float16)(O[n][v] * inv);
  }
}

// ---------------- SiLU(gate) * up -> f16 ----------------
__global__ void k_silu_mul(const float* __restrict__ g, const float* __restrict__ u,
                           _Float16* __restrict__ mid, int total) {
  int i = blockIdx.x * blockDim.x + threadIdx.x;
  if (i >= total) return;
  float x = g[i];
  float s = x / (1.f + __expf(-x));
  mid[i] = (_Float16)(s * u[i]);
}

// ============================================================================
extern "C" void kernel_launch(void* const* d_in, const int* in_sizes, int n_in,
                              void* d_out, int out_size, void* d_ws, size_t ws_size,
                              hipStream_t stream) {
  const float* x      = (const float*)d_in[0];
  const float* mask   = (const float*)d_in[1];
  const float* rope   = (const float*)d_in[2];
  const float* n1w    = (const float*)d_in[3];
  const float* n2w    = (const float*)d_in[4];
  const float* qkv_w  = (const float*)d_in[5];
  const float* qkv_b  = (const float*)d_in[6];
  const float* proj_w = (const float*)d_in[7];
  const float* proj_b = (const float*)d_in[8];
  const float* gate_w = (const float*)d_in[9];
  const float* gate_b = (const float*)d_in[10];
  const float* up_w   = (const float*)d_in[11];
  const float* up_b   = (const float*)d_in[12];
  const float* down_w = (const float*)d_in[13];
  const float* down_b = (const float*)d_in[14];
  float* out = (float*)d_out;

  char* ws = (char*)d_ws;
  size_t off = 0;
  auto alloc = [&](size_t bytes) -> void* {
    void* p = ws + off;
    off = (off + bytes + 255) & ~(size_t)255;
    return p;
  };

  _Float16* h16     = (_Float16*)alloc((size_t)SEQ * DIM * 2);
  _Float16* h2_16   = (_Float16*)alloc((size_t)SEQ * DIM * 2);
  _Float16* qkvw16  = (_Float16*)alloc((size_t)QKVN * DIM * 2);
  _Float16* projw16 = (_Float16*)alloc((size_t)DIM * DIM * 2);
  _Float16* gatew16 = (_Float16*)alloc((size_t)FFNP * DIM * 2);
  _Float16* upw16   = (_Float16*)alloc((size_t)FFNP * DIM * 2);
  _Float16* downw16 = (_Float16*)alloc((size_t)DIM * FFNP * 2);
  float*    qkvf32  = (float*)alloc((size_t)SEQ * QKVN * 4);
  _Float16* q96     = (_Float16*)alloc((size_t)HEADS * SEQ * HDP * 2);
  _Float16* k96     = (_Float16*)alloc((size_t)HEADS * SEQ * HDP * 2);
  _Float16* vt      = (_Float16*)alloc((size_t)HEADS * HD * SEQ * 2);
  _Float16* attn16  = (_Float16*)alloc((size_t)SEQ * DIM * 2);
  float*    x2      = (float*)alloc((size_t)SEQ * DIM * 4);
  float*    gatef   = (float*)alloc((size_t)SEQ * FFNP * 4);
  float*    upf     = (float*)alloc((size_t)SEQ * FFNP * 4);
  _Float16* mid16   = (_Float16*)alloc((size_t)SEQ * FFNP * 2);

  auto cvt = [&](const float* s, _Float16* d, int sr, int sc, int dr, int dc) {
    int tot = dr * dc;
    k_cvt_pad<<<(tot + 255) / 256, 256, 0, stream>>>(s, d, sr, sc, dr, dc);
  };

  cvt(qkv_w, qkvw16, QKVN, DIM, QKVN, DIM);
  cvt(proj_w, projw16, DIM, DIM, DIM, DIM);
  cvt(gate_w, gatew16, FFN, DIM, FFNP, DIM);
  cvt(up_w, upw16, FFN, DIM, FFNP, DIM);
  cvt(down_w, downw16, DIM, FFN, DIM, FFNP);

  // 1) h = rmsnorm(x, n1w)
  k_rmsnorm<<<SEQ, 256, 0, stream>>>(x, n1w, h16);

  // 2) qkv = h @ qkv_w^T + qkv_b
  k_gemm<false><<<dim3(QKVN / 128, SEQ / 128), 256, 0, stream>>>(
      h16, qkvw16, qkv_b, QKVN, nullptr, qkvf32, SEQ, QKVN, DIM);

  // 3) rope + repack
  k_rope_pack<<<SEQ, 256, 0, stream>>>(qkvf32, rope, q96, k96, vt);

  // 4) attention (flash, fused softmax)
  k_flash<<<dim3(SEQ / 64, HEADS), 128, 0, stream>>>(q96, k96, vt, mask, attn16);

  // 5) x2 = attn @ proj_w^T + proj_b + x
  k_gemm<true><<<dim3(DIM / 128, SEQ / 128), 256, 0, stream>>>(
      attn16, projw16, proj_b, DIM, x, x2, SEQ, DIM, DIM);

  // 6) h2 = rmsnorm(x2, n2w)
  k_rmsnorm<<<SEQ, 256, 0, stream>>>(x2, n2w, h2_16);

  // 7) gate / up GEMMs (padded N = FFNP; pad cols get zero weights + zero bias)
  k_gemm<false><<<dim3(FFNP / 128, SEQ / 128), 256, 0, stream>>>(
      h2_16, gatew16, gate_b, FFN, nullptr, gatef, SEQ, FFNP, DIM);
  k_gemm<false><<<dim3(FFNP / 128, SEQ / 128), 256, 0, stream>>>(
      h2_16, upw16, up_b, FFN, nullptr, upf, SEQ, FFNP, DIM);

  // 8) mid = silu(gate) * up
  {
    int tot = SEQ * FFNP;
    k_silu_mul<<<(tot + 255) / 256, 256, 0, stream>>>(gatef, upf, mid16, tot);
  }

  // 9) out = mid @ down_w^T + down_b + x2
  k_gemm<true><<<dim3(DIM / 128, SEQ / 128), 256, 0, stream>>>(
      mid16, downw16, down_b, DIM, x2, out, SEQ, DIM, FFNP);
}
